// FrequencyDCTBranch_70385924046877
// MI455X (gfx1250) — matrix operand order repaired and checked by reading, not compiled, not run on hardware
//
#include <hip/hip_runtime.h>
#include <hip/hip_bf16.h>
#include <math.h>

typedef __attribute__((ext_vector_type(16))) _Float16 v16h;
typedef __attribute__((ext_vector_type(8)))  float    v8f;

#define PATCH 8
#define HW_DIM 224
#define NPH 28
#define NPATCH 784   // 28*28
#define FRAMES 256   // B*T
#define FEAT 192     // 3*64
#define D_IN 384
#define HID 512
#define D_OUT 256

// ---------------------------------------------------------------------------
// Stage 1: single fused streaming pass over x_raw (one block per frame).
//   grayscale (unscaled) -> 8x8 patchify -> 2D DCT -> per-coefficient
//   Sum / SumSq accumulated in *registers* per thread, plus raw per-frame max.
//   Block-end: conflict-free staged LDS reduction (no atomics).
// Rescale by 255 is applied later (linear through the DCT).
// ---------------------------------------------------------------------------
__global__ void __launch_bounds__(256)
frame_pass_kernel(const float* __restrict__ x,
                  float* __restrict__ frameS,   // [256][64] unscaled sum
                  float* __restrict__ frameQ,   // [256][64] unscaled sumsq
                  float* __restrict__ frameMax) // [256] raw max
{
    __shared__ float Ds[8][8];          // DCT-II matrix
    __shared__ float red[256 * 33];     // stride-33 padding -> no bank conflicts

    const int tid = threadIdx.x;
    if (tid < 64) {
        int k = tid >> 3, n = tid & 7;
        float alpha = (k == 0) ? 0.35355339059327373f : 0.5f; // 1/sqrt(8), sqrt(2/8)
        Ds[k][n] = alpha * cosf(3.14159265358979323846f * (2.0f * n + 1.0f) * k / 16.0f);
    }
    __syncthreads();

    const int frame = blockIdx.x;
    const float* base = x + (size_t)frame * 3u * HW_DIM * HW_DIM;
    const int HW = HW_DIM * HW_DIM;

    float accS[64], accQ[64];
    #pragma unroll
    for (int c = 0; c < 64; ++c) { accS[c] = 0.0f; accQ[c] = 0.0f; }
    float pmax = 0.0f;

    for (int p = tid; p < NPATCH; p += 256) {
        int ph = p / NPH, pw = p % NPH;
        float t[8][8];
        #pragma unroll
        for (int i = 0; i < 8; ++i) {
            int idx = (ph * PATCH + i) * HW_DIM + pw * PATCH;  // multiple of 8 -> 32B aligned
            // vectorized b128 loads: 2x float4 per channel
            float4 r0 = *(const float4*)(base + idx);
            float4 r1 = *(const float4*)(base + idx + 4);
            float4 g0 = *(const float4*)(base + HW + idx);
            float4 g1 = *(const float4*)(base + HW + idx + 4);
            float4 b0 = *(const float4*)(base + 2 * HW + idx);
            float4 b1 = *(const float4*)(base + 2 * HW + idx + 4);
            pmax = fmaxf(pmax, fmaxf(fmaxf(fmaxf(r0.x, r0.y), fmaxf(r0.z, r0.w)),
                                     fmaxf(fmaxf(r1.x, r1.y), fmaxf(r1.z, r1.w))));
            pmax = fmaxf(pmax, fmaxf(fmaxf(fmaxf(g0.x, g0.y), fmaxf(g0.z, g0.w)),
                                     fmaxf(fmaxf(g1.x, g1.y), fmaxf(g1.z, g1.w))));
            pmax = fmaxf(pmax, fmaxf(fmaxf(fmaxf(b0.x, b0.y), fmaxf(b0.z, b0.w)),
                                     fmaxf(fmaxf(b1.x, b1.y), fmaxf(b1.z, b1.w))));
            float row[8];
            row[0] = 0.299f * r0.x + 0.587f * g0.x + 0.114f * b0.x;
            row[1] = 0.299f * r0.y + 0.587f * g0.y + 0.114f * b0.y;
            row[2] = 0.299f * r0.z + 0.587f * g0.z + 0.114f * b0.z;
            row[3] = 0.299f * r0.w + 0.587f * g0.w + 0.114f * b0.w;
            row[4] = 0.299f * r1.x + 0.587f * g1.x + 0.114f * b1.x;
            row[5] = 0.299f * r1.y + 0.587f * g1.y + 0.114f * b1.y;
            row[6] = 0.299f * r1.z + 0.587f * g1.z + 0.114f * b1.z;
            row[7] = 0.299f * r1.w + 0.587f * g1.w + 0.114f * b1.w;
            // t[i][l] = sum_j row[j] * Ds[l][j]   (row transform)
            #pragma unroll
            for (int l = 0; l < 8; ++l) {
                float v = 0.0f;
                #pragma unroll
                for (int j = 0; j < 8; ++j) v = fmaf(row[j], Ds[l][j], v);
                t[i][l] = v;
            }
        }
        // dct[k][l] = sum_i Ds[k][i] * t[i][l]; accumulate per-coefficient stats
        #pragma unroll
        for (int k = 0; k < 8; ++k) {
            #pragma unroll
            for (int l = 0; l < 8; ++l) {
                float v = 0.0f;
                #pragma unroll
                for (int i = 0; i < 8; ++i) v = fmaf(Ds[k][i], t[i][l], v);
                accS[k * 8 + l] += v;
                accQ[k * 8 + l] = fmaf(v, v, accQ[k * 8 + l]);
            }
        }
    }

    // Staged block reduction: 4 chunks of 32 values, no atomics, no conflicts.
    #pragma unroll
    for (int chunk = 0; chunk < 4; ++chunk) {
        #pragma unroll
        for (int c = 0; c < 32; ++c) {
            float v = (chunk == 0) ? accS[c] :
                      (chunk == 1) ? accS[32 + c] :
                      (chunk == 2) ? accQ[c] : accQ[32 + c];
            red[tid * 33 + c] = v;
        }
        __syncthreads();
        if (tid < 32) {
            float s = 0.0f;
            for (int t = 0; t < 256; ++t) s += red[t * 33 + tid];
            int coeff = (chunk & 1) * 32 + tid;
            if (chunk < 2) frameS[frame * 64 + coeff] = s;
            else           frameQ[frame * 64 + coeff] = s;
        }
        __syncthreads();
    }

    // per-frame raw max
    red[tid] = pmax;
    __syncthreads();
    for (int s = 128; s > 0; s >>= 1) {
        if (tid < s) red[tid] = fmaxf(red[tid], red[tid + s]);
        __syncthreads();
    }
    if (tid == 0) frameMax[frame] = red[0];
}

// ---------------------------------------------------------------------------
// Stage 2: reduce 256 per-frame maxima -> scale (255 or 1)
// ---------------------------------------------------------------------------
__global__ void scale_kernel(const float* __restrict__ frameMax, float* __restrict__ scale) {
    __shared__ float red[256];
    int tid = threadIdx.x;
    red[tid] = frameMax[tid];
    __syncthreads();
    for (int s = 128; s > 0; s >>= 1) {
        if (tid < s) red[tid] = fmaxf(red[tid], red[tid + s]);
        __syncthreads();
    }
    if (tid == 0) scale[0] = (red[0] <= 1.01f) ? 255.0f : 1.0f;
}

// ---------------------------------------------------------------------------
// Stage 3: apply scale, produce per-frame features [256][192]
//   mu = c*S/N; sigma = sqrt((c^2*Q - (c*S)^2/N)/(N-1)); energy = log1p(c^2*Q/N)
// ---------------------------------------------------------------------------
__global__ void finalize_kernel(const float* __restrict__ frameS,
                                const float* __restrict__ frameQ,
                                const float* __restrict__ scale_p,
                                float* __restrict__ ff) {
    int frame = blockIdx.x;
    int c = threadIdx.x;           // 64 threads
    float sc = scale_p[0];
    const float N = (float)NPATCH;
    float s = frameS[frame * 64 + c] * sc;
    float q = frameQ[frame * 64 + c] * sc * sc;
    float mu = s / N;
    float var = (q - s * s / N) / (N - 1.0f);
    var = fmaxf(var, 0.0f);
    float* o = ff + (size_t)frame * FEAT;
    o[c]        = mu;
    o[64 + c]   = sqrtf(var);
    o[128 + c]  = log1pf(q / N);
}

// ---------------------------------------------------------------------------
// Stage 4: temporal mean/std(ddof=1) over T=16 -> freq [16,384] stored as f16
// ---------------------------------------------------------------------------
__global__ void temporal_kernel(const float* __restrict__ ff, _Float16* __restrict__ freqh) {
    int gid = blockIdx.x * blockDim.x + threadIdx.x;
    if (gid >= 16 * FEAT) return;
    int b = gid / FEAT, j = gid % FEAT;
    float s = 0.0f, q = 0.0f;
    #pragma unroll
    for (int t = 0; t < 16; ++t) {
        float v = ff[(size_t)(b * 16 + t) * FEAT + j];
        s += v;
        q += v * v;
    }
    float mu = s / 16.0f;
    float var = (q - s * s / 16.0f) / 15.0f;
    var = fmaxf(var, 0.0f);
    freqh[b * D_IN + j]        = (_Float16)mu;
    freqh[b * D_IN + FEAT + j] = (_Float16)sqrtf(var);
}

// ---------------------------------------------------------------------------
// WMMA helpers (wave32, v_wmma_f32_16x16x32_f16)
// ---------------------------------------------------------------------------
__device__ __forceinline__ float gelu_exact(float v) {
    return 0.5f * v * (1.0f + erff(v * 0.70710678118654752f));
}

// A fragment: 16xK f16 row-major; lane row = lane&15; K split 0/8 with +16 jump at VGPR4
__device__ __forceinline__ v16h load_a_frag(const _Float16* __restrict__ A, int lda,
                                            int k0, int lane) {
    int row = lane & 15;
    int kb = (lane >> 4) ? 8 : 0;
    v16h a;
    #pragma unroll
    for (int e = 0; e < 16; ++e) {
        int vg = e >> 1, h = e & 1;
        int k = (vg < 4) ? (kb + 2 * vg + h) : (kb + 16 + 2 * (vg - 4) + h);
        a[e] = A[row * lda + k0 + k];
    }
    return a;
}

// B fragment: 32x16 tile of row-major f32 weights, converted to f16 on the fly.
// lane col = lane&15; lanes 0-15 hold K=k0+{0..7,16..23}, lanes 16-31 hold K=k0+{8..15,24..31}.
__device__ __forceinline__ v16h load_b_frag(const float* __restrict__ W, int ldn,
                                            int k0, int n0, int lane) {
    int col = lane & 15;
    int kb = (lane >> 4) ? 8 : 0;
    v16h b;
    #pragma unroll
    for (int e = 0; e < 16; ++e)
        b[e] = (_Float16)W[(size_t)(k0 + kb + ((e < 8) ? e : e + 8)) * ldn + n0 + col];
    return b;
}

// ---------------------------------------------------------------------------
// Stage 5: layer 1  [16,384] x [384,512] + b1, exact GELU -> f16 [16,512]
// One wave per 16x16 output tile; 12 WMMA k-steps.
// ---------------------------------------------------------------------------
__global__ void mlp1_kernel(const _Float16* __restrict__ freqh,
                            const float* __restrict__ W1,
                            const float* __restrict__ b1,
                            _Float16* __restrict__ hh) {
    int lane = threadIdx.x;           // blockDim.x == 32
    int n0 = blockIdx.x * 16;
    v8f acc = {};
    for (int k0 = 0; k0 < D_IN; k0 += 32) {
        v16h a = load_a_frag(freqh, D_IN, k0, lane);
        v16h b = load_b_frag(W1, HID, k0, n0, lane);
        acc = __builtin_amdgcn_wmma_f32_16x16x32_f16(false, a, false, b,
                                                     (short)0, acc, false, false);
    }
    int col = lane & 15;
    int ro = (lane >> 4) ? 8 : 0;
    #pragma unroll
    for (int r = 0; r < 8; ++r) {
        int m = r + ro;
        int n = n0 + col;
        hh[m * HID + n] = (_Float16)gelu_exact(acc[r] + b1[n]);
    }
}

// ---------------------------------------------------------------------------
// Stage 6: layer 2  [16,512] x [512,256] + b2, exact GELU -> f32 out
// 16 WMMA k-steps.
// ---------------------------------------------------------------------------
__global__ void mlp2_kernel(const _Float16* __restrict__ hh,
                            const float* __restrict__ W2,
                            const float* __restrict__ b2,
                            float* __restrict__ out) {
    int lane = threadIdx.x;           // blockDim.x == 32
    int n0 = blockIdx.x * 16;
    v8f acc = {};
    for (int k0 = 0; k0 < HID; k0 += 32) {
        v16h a = load_a_frag(hh, HID, k0, lane);
        v16h b = load_b_frag(W2, D_OUT, k0, n0, lane);
        acc = __builtin_amdgcn_wmma_f32_16x16x32_f16(false, a, false, b,
                                                     (short)0, acc, false, false);
    }
    int col = lane & 15;
    int ro = (lane >> 4) ? 8 : 0;
    #pragma unroll
    for (int r = 0; r < 8; ++r) {
        int m = r + ro;
        int n = n0 + col;
        out[m * D_OUT + n] = gelu_exact(acc[r] + b2[n]);
    }
}

// ---------------------------------------------------------------------------
// Launch
// ---------------------------------------------------------------------------
extern "C" void kernel_launch(void* const* d_in, const int* in_sizes, int n_in,
                              void* d_out, int out_size, void* d_ws, size_t ws_size,
                              hipStream_t stream) {
    const float* x_raw = (const float*)d_in[0];
    const float* W1    = (const float*)d_in[1];
    const float* b1    = (const float*)d_in[2];
    const float* W2    = (const float*)d_in[3];
    const float* b2    = (const float*)d_in[4];
    float* out = (float*)d_out;

    // Workspace layout (byte offsets)
    char* ws = (char*)d_ws;
    float*    d_frameS   = (float*)(ws + 0);        // 256*64 f32 = 65536 B
    float*    d_frameQ   = (float*)(ws + 65536);    // 256*64 f32 = 65536 B
    float*    d_frameMax = (float*)(ws + 131072);   // 256 f32
    float*    d_scale    = (float*)(ws + 132096);   // 1 f32
    float*    d_ff       = (float*)(ws + 133120);   // 256*192 f32 = 196608 B
    _Float16* d_freqh    = (_Float16*)(ws + 330752);// 16*384 f16 = 12288 B
    _Float16* d_hh       = (_Float16*)(ws + 343040);// 16*512 f16 = 16384 B

    // 1) single streaming pass: gray + DCT + unscaled stats + raw max
    frame_pass_kernel<<<FRAMES, 256, 0, stream>>>(x_raw, d_frameS, d_frameQ, d_frameMax);

    // 2) rescale decision from per-frame maxima
    scale_kernel<<<1, 256, 0, stream>>>(d_frameMax, d_scale);

    // 3) apply scale -> per-frame features [256][192]
    finalize_kernel<<<FRAMES, 64, 0, stream>>>(d_frameS, d_frameQ, d_scale, d_ff);

    // 4) temporal mean/std -> freq [16,384] in f16
    temporal_kernel<<<(16 * FEAT + 255) / 256, 256, 0, stream>>>(d_ff, d_freqh);

    // 5) MLP layer 1 via WMMA: 32 tiles of 16x16 over N=512
    mlp1_kernel<<<HID / 16, 32, 0, stream>>>(d_freqh, W1, b1, d_hh);

    // 6) MLP layer 2 via WMMA: 16 tiles over N=256 -> f32 output
    mlp2_kernel<<<D_OUT / 16, 32, 0, stream>>>(d_hh, W2, b2, out);
}
